// CNLinkPredictor_13657996001674
// MI455X (gfx1250) — matrix operand order repaired and checked by reading, not compiled, not run on hardware
//
#include <hip/hip_runtime.h>

// ---------------- types ----------------
typedef __bf16 bf16_t;
typedef __attribute__((ext_vector_type(16))) __bf16 v16bf;
typedef __attribute__((ext_vector_type(8)))  float  v8f;
typedef __attribute__((ext_vector_type(4)))  unsigned int u32x4;

union FragAB { v16bf v; u32x4 q[2]; };

// ---------------- problem constants ----------------
#define N_NODES 10000
#define IN_CH   128
#define HID     256
#define N_EDGES 320000
#define E_TGT   8192
#define WPR     320   // padded u32 words per bitset row (313 used)
#define WUSED   313

// ---------------- bitset adjacency ----------------
__global__ void cnlp_zero_u32(unsigned int* p, int n) {
    int t = blockIdx.x * blockDim.x + threadIdx.x;
    if (t < n) p[t] = 0u;
}

__global__ void cnlp_scatter_bits(const int* __restrict__ row,
                                  const int* __restrict__ col,
                                  unsigned int* __restrict__ bs, int nEdges) {
    int t = blockIdx.x * blockDim.x + threadIdx.x;
    if (t >= nEdges) return;
    int r = row[t], c = col[t];
    atomicOr(&bs[(size_t)r * WPR + (c >> 5)], 1u << (c & 31));
}

// ---------------- weight transpose + bf16 convert: Wt[n*K+k] = W[k*N+n] ----------------
__global__ void cnlp_wt_bf16(const float* __restrict__ W, bf16_t* __restrict__ Wt,
                             int K, int N) {
    int t = blockIdx.x * blockDim.x + threadIdx.x;
    if (t >= K * N) return;
    int k = t / N, n = t - k * N;
    Wt[(size_t)n * K + k] = (bf16_t)W[t];
}

// ---------------- edge features: cn_x and xi*xj (one wave per edge) ----------------
__global__ void cnlp_edge_feat(const float* __restrict__ x,
                               const unsigned int* __restrict__ bs,
                               const int* __restrict__ tar,   // [2*E]
                               bf16_t* __restrict__ cnA,      // [E, IN_CH] bf16
                               bf16_t* __restrict__ pijA,     // [E, IN_CH] bf16
                               int E) {
    int wv = (blockIdx.x * blockDim.x + threadIdx.x) >> 5;
    int lane = threadIdx.x & 31;
    if (wv >= E) return;
    int i = tar[wv], j = tar[E + wv];
    const unsigned int* ri = bs + (size_t)i * WPR;
    const unsigned int* rj = bs + (size_t)j * WPR;
    float a0 = 0.f, a1 = 0.f, a2 = 0.f, a3 = 0.f;
    for (int w = 0; w < WUSED; ++w) {
        unsigned int aw = ri[w] & rj[w];   // uniform across the wave
        while (aw) {
            int b = __builtin_ctz(aw);
            aw &= aw - 1u;
            const float* xk = x + (size_t)(w * 32 + b) * IN_CH;
            a0 += xk[lane];
            a1 += xk[lane + 32];
            a2 += xk[lane + 64];
            a3 += xk[lane + 96];
        }
    }
    const float* xi = x + (size_t)i * IN_CH;
    const float* xj = x + (size_t)j * IN_CH;
    size_t o = (size_t)wv * IN_CH;
    cnA[o + lane     ] = (bf16_t)a0;
    cnA[o + lane + 32] = (bf16_t)a1;
    cnA[o + lane + 64] = (bf16_t)a2;
    cnA[o + lane + 96] = (bf16_t)a3;
    pijA[o + lane     ] = (bf16_t)(xi[lane     ] * xj[lane     ]);
    pijA[o + lane + 32] = (bf16_t)(xi[lane + 32] * xj[lane + 32]);
    pijA[o + lane + 64] = (bf16_t)(xi[lane + 64] * xj[lane + 64]);
    pijA[o + lane + 96] = (bf16_t)(xi[lane + 96] * xj[lane + 96]);
}

// ---------------- WMMA GEMM: C[M,N] = A[M,K](bf16) * Bt[N,K]^T(bf16) + bias
//                  + (optional) beta*addend ; optional ReLU; bf16 and/or f32 out.
//                  One wave per 16x16 tile.  ----------------
__global__ void __launch_bounds__(256)
cnlp_wmma_gemm(const bf16_t* __restrict__ A,
               const bf16_t* __restrict__ Bt,
               const float* __restrict__ bias,
               const float* __restrict__ addend,   // may be null; [M,N] f32
               const float* __restrict__ betaPtr,  // may be null; scalar
               bf16_t* __restrict__ outBf,         // may be null
               float*  __restrict__ outF,          // may be null
               int M, int N, int K, int doRelu) {
    int wave = (blockIdx.x * blockDim.x + threadIdx.x) >> 5;
    int lane = threadIdx.x & 31;
    int tilesN = N >> 4;
    int tm = wave / tilesN;
    int tn = wave - tm * tilesN;
    if (tm * 16 >= M) return;

    int row16 = lane & 15;
    int half  = lane >> 4;

    const bf16_t* aRow = A  + (size_t)(tm * 16 + row16) * K;
    const bf16_t* bRow = Bt + (size_t)(tn * 16 + row16) * K;

    v8f acc = {};
    for (int kk = 0; kk < K; kk += 32) {
        FragAB a, b;
        // A 16x32 bf16: lane holds K = kk+half*8 .. +7  and  kk+16+half*8 .. +7
        a.q[0] = *(const u32x4*)(aRow + kk + half * 8);
        a.q[1] = *(const u32x4*)(aRow + kk + 16 + half * 8);
        // B 32x16 bf16 (column n = row of Bt): K = kk+half*8 .. +7  and  kk+16+half*8 .. +7
        b.q[0] = *(const u32x4*)(bRow + kk + half * 8);
        b.q[1] = *(const u32x4*)(bRow + kk + 16 + half * 8);
        acc = __builtin_amdgcn_wmma_f32_16x16x32_bf16(
            false, a.v, false, b.v, (short)0, acc, false, false);
    }

    int n = tn * 16 + row16;
    float bn  = bias ? bias[n] : 0.f;
    float bet = betaPtr ? betaPtr[0] : 0.f;
    #pragma unroll
    for (int v = 0; v < 8; ++v) {
        int m = tm * 16 + v + half * 8;
        float val = acc[v] + bn;
        if (addend) val += bet * addend[(size_t)m * N + n];
        if (doRelu) val = val > 0.f ? val : 0.f;
        size_t idx = (size_t)m * N + n;
        if (outBf) outBf[idx] = (bf16_t)val;
        if (outF)  outF[idx]  = val;
    }
}

// ---------------- final head GEMV: out[e] = h[e,:]·wl2 + bl2 ----------------
__global__ void cnlp_head_gemv(const float* __restrict__ h,
                               const float* __restrict__ wl2,
                               const float* __restrict__ bl2,
                               float* __restrict__ out, int E) {
    int wv = (blockIdx.x * blockDim.x + threadIdx.x) >> 5;
    int lane = threadIdx.x & 31;
    if (wv >= E) return;
    const float* hr = h + (size_t)wv * HID;
    float s = 0.f;
    #pragma unroll
    for (int c = 0; c < HID; c += 32) s += hr[c + lane] * wl2[c + lane];
    #pragma unroll
    for (int off = 16; off > 0; off >>= 1) s += __shfl_xor(s, off, 32);
    if (lane == 0) out[wv] = s + bl2[0];
}

// ---------------- host launcher ----------------
static inline void launch_gemm(hipStream_t stream,
                               const bf16_t* A, const bf16_t* Bt, const float* bias,
                               const float* addend, const float* beta,
                               bf16_t* outBf, float* outF,
                               int M, int N, int K, int relu) {
    int tiles = (M / 16) * (N / 16);
    int blocks = (tiles + 7) / 8;  // 8 waves (16x16 tiles) per 256-thread block
    cnlp_wmma_gemm<<<blocks, 256, 0, stream>>>(A, Bt, bias, addend, beta,
                                               outBf, outF, M, N, K, relu);
}

extern "C" void kernel_launch(void* const* d_in, const int* in_sizes, int n_in,
                              void* d_out, int out_size, void* d_ws, size_t ws_size,
                              hipStream_t stream) {
    const float* x       = (const float*)d_in[0];
    const int*   adj_row = (const int*)  d_in[1];
    const int*   adj_col = (const int*)  d_in[2];
    const int*   tar_ei  = (const int*)  d_in[3];
    const float* beta    = (const float*)d_in[4];
    const float* W1   = (const float*)d_in[5];  const float* b1   = (const float*)d_in[6];
    const float* W2   = (const float*)d_in[7];  const float* b2   = (const float*)d_in[8];
    const float* W3   = (const float*)d_in[9];  const float* b3   = (const float*)d_in[10];
    const float* Wij1 = (const float*)d_in[11]; const float* bij1 = (const float*)d_in[12];
    const float* Wij2 = (const float*)d_in[13]; const float* bij2 = (const float*)d_in[14];
    const float* Wl1  = (const float*)d_in[15]; const float* bl1  = (const float*)d_in[16];
    const float* Wl2  = (const float*)d_in[17]; const float* bl2  = (const float*)d_in[18];
    float* out = (float*)d_out;

    const int E = E_TGT;
    char* base = (char*)d_ws;
    size_t off = 0;
    auto take = [&](size_t bytes) -> char* {
        char* p = base + off;
        off += (bytes + 255) & ~(size_t)255;
        return p;
    };
    unsigned int* bs  = (unsigned int*)take((size_t)N_NODES * WPR * 4);
    bf16_t* cnA  = (bf16_t*)take((size_t)E * IN_CH * 2);
    bf16_t* pijA = (bf16_t*)take((size_t)E * IN_CH * 2);
    bf16_t* tA   = (bf16_t*)take((size_t)E * HID * 2);
    bf16_t* tB   = (bf16_t*)take((size_t)E * HID * 2);
    float*  xcn  = (float*) take((size_t)E * HID * 4);
    bf16_t* zA   = (bf16_t*)take((size_t)E * HID * 2);
    float*  h    = (float*) take((size_t)E * HID * 4);
    bf16_t* W1t   = (bf16_t*)take((size_t)IN_CH * HID * 2);
    bf16_t* Wij1t = (bf16_t*)take((size_t)IN_CH * HID * 2);
    bf16_t* W2t   = (bf16_t*)take((size_t)HID * HID * 2);
    bf16_t* W3t   = (bf16_t*)take((size_t)HID * HID * 2);
    bf16_t* Wij2t = (bf16_t*)take((size_t)HID * HID * 2);
    bf16_t* Wl1t  = (bf16_t*)take((size_t)HID * HID * 2);

    // 1) binary adjacency bitset (duplicates collapse, matching reference)
    int words = N_NODES * WPR;
    cnlp_zero_u32<<<(words + 255) / 256, 256, 0, stream>>>(bs, words);
    cnlp_scatter_bits<<<(N_EDGES + 255) / 256, 256, 0, stream>>>(adj_row, adj_col, bs, N_EDGES);

    // 2) transpose + bf16-convert weights (Bt layout [N,K])
    cnlp_wt_bf16<<<(IN_CH * HID + 255) / 256, 256, 0, stream>>>(W1,   W1t,   IN_CH, HID);
    cnlp_wt_bf16<<<(IN_CH * HID + 255) / 256, 256, 0, stream>>>(Wij1, Wij1t, IN_CH, HID);
    cnlp_wt_bf16<<<(HID * HID + 255) / 256, 256, 0, stream>>>(W2,   W2t,   HID, HID);
    cnlp_wt_bf16<<<(HID * HID + 255) / 256, 256, 0, stream>>>(W3,   W3t,   HID, HID);
    cnlp_wt_bf16<<<(HID * HID + 255) / 256, 256, 0, stream>>>(Wij2, Wij2t, HID, HID);
    cnlp_wt_bf16<<<(HID * HID + 255) / 256, 256, 0, stream>>>(Wl1,  Wl1t,  HID, HID);

    // 3) per-edge common-neighbor sum + elementwise product (one wave per edge)
    cnlp_edge_feat<<<(E * 32 + 255) / 256, 256, 0, stream>>>(x, bs, tar_ei, cnA, pijA, E);

    // 4) MLP chain via WMMA bf16 GEMMs
    // xcn branch: relu(cn@W1+b1) -> relu(@W2+b2) -> @W3+b3
    launch_gemm(stream, cnA, W1t, b1, nullptr, nullptr, tA, nullptr, E, HID, IN_CH, 1);
    launch_gemm(stream, tA,  W2t, b2, nullptr, nullptr, tB, nullptr, E, HID, HID,   1);
    launch_gemm(stream, tB,  W3t, b3, nullptr, nullptr, nullptr, xcn, E, HID, HID,  0);
    // xij branch: relu(pij@Wij1+bij1) -> @Wij2+bij2, fused z = xij + beta*xcn
    launch_gemm(stream, pijA, Wij1t, bij1, nullptr, nullptr, tA, nullptr, E, HID, IN_CH, 1);
    launch_gemm(stream, tA,   Wij2t, bij2, xcn, beta, zA, nullptr, E, HID, HID, 0);
    // head: h = relu(z@Wl1+bl1)
    launch_gemm(stream, zA, Wl1t, bl1, nullptr, nullptr, nullptr, h, E, HID, HID, 1);

    // 5) out = h @ Wl2 + bl2   (256 -> 1 GEMV, one wave per edge)
    cnlp_head_gemv<<<(E * 32 + 255) / 256, 256, 0, stream>>>(h, Wl2, bl2, out, E);
}
